// ManifoldNeuralSDE_67267777789992
// MI455X (gfx1250) — matrix operand order repaired and checked by reading, not compiled, not run on hardware
//
#include <hip/hip_runtime.h>

// ---------------- problem constants ----------------
#define B_TOTAL 4096
#define T_STEPS 32
#define SDIM    25     // 5*5 flattened state
#define CDIM    64     // context dim
#define HID     128
#define DDIM    15     // SPD tangent dim
#define NSTEPS  5
#define DT_C    0.2f

// ---------------- WMMA types / helpers ----------------
typedef __attribute__((ext_vector_type(16))) __bf16 v16bf;
typedef __attribute__((ext_vector_type(8)))  float  v8f;
typedef __attribute__((ext_vector_type(4)))  unsigned int u32x4;
typedef __attribute__((ext_vector_type(8)))  unsigned int u32x8;

__device__ __forceinline__ v8f wmma_bf16(v16bf a, v16bf b, v8f c) {
  // D = A(16x32) * B(32x16) + C, f32 accumulate
  return __builtin_amdgcn_wmma_f32_16x16x32_bf16(
      /*neg_a=*/false, a, /*neg_b=*/false, b,
      /*c_mod=*/(short)0, c, /*reuse_a=*/false, /*reuse_b=*/false);
}

// A-matrix 16x32 bf16 fragment from a row-major bf16 buffer (row stride ld elems).
// ISA layout: lane m=lane&15; lanes 0-15: K in [0,8) U [16,24); lanes 16-31: +8.
__device__ __forceinline__ v16bf load_a_frag(const __bf16* buf, int ld, int kc) {
  const int lane  = threadIdx.x & 31;
  const int m     = lane & 15;
  const int kbase = kc * 32 + ((lane & 16) ? 8 : 0);
  const __bf16* row = buf + m * ld + kbase;
  v16bf a;
#pragma unroll
  for (int e = 0; e < 8; e++) a[e] = row[e];
#pragma unroll
  for (int e = 0; e < 8; e++) a[8 + e] = row[16 + e];
  return a;
}

// B fragments pre-packed so each lane loads 16 contiguous bf16 (32B) from LDS.
__device__ __forceinline__ v16bf load_b_frag(const __bf16* packed, int nt, int kt, int KT) {
  const int lane = threadIdx.x & 31;
  return *(const v16bf*)(packed + (((size_t)nt * KT + kt) * 32 + lane) * 16);
}

// ---------------- fast transcendental helpers (single hardware op paths) ----------------
__device__ __forceinline__ float frcp_(float x)     { return __builtin_amdgcn_rcpf(x); }
__device__ __forceinline__ float fsqrt_(float x)    { return __builtin_amdgcn_sqrtf(x); }
__device__ __forceinline__ float sigmoidf_(float x) { return frcp_(1.f + __expf(-x)); }
__device__ __forceinline__ float siluf_(float x)    { return x * sigmoidf_(x); }
__device__ __forceinline__ float softplusf_(float x){
  return (x > 20.f) ? x : __logf(1.f + __expf(x));
}
__device__ __forceinline__ float tanhf_(float x) {
  float e = __expf(-2.f * fabsf(x));
  float r = (1.f - e) * frcp_(1.f + e);
  return copysignf(r, x);
}

// ---------------- Tensor Data Mover: 1-D global -> LDS DMA ----------------
// D# per CDNA5 ISA: group0 {count=1, lds_addr, global_addr, type=2},
// group1 {data_size=8B, tensor_dim0=nd, tensor_dim1=1, tile_dim0=nd, tile_dim1=1,
//         tensor_dim0_stride=nd}. Groups 2/3 zero (dims unused). Tracked by TENSORcnt.
__device__ __forceinline__ void tdm_load_1d(const void* g, void* l, unsigned bytes) {
  unsigned long long ga = (unsigned long long)(size_t)g;
  unsigned lo = (unsigned)(size_t)l;          // flat LDS addr low 32 bits = LDS offset
  unsigned nd = bytes >> 3;                   // 8-byte elements (tile_dim0 <= 65535)
  u32x4 g0 = { 1u,                            // count=1 (user descriptor)
               lo,                            // lds_addr
               (unsigned)(ga & 0xFFFFFFFFu),  // global_addr[31:0]
               ((unsigned)((ga >> 32) & 0x01FFFFFFu)) | (2u << 30) }; // addr[56:32]|type=2
  u32x8 g1 = { (3u << 16),                          // data_size = 8B
               (nd & 0xFFFFu) << 16,                // tensor_dim0[15:0]  @bits 63:48
               ((nd >> 16) & 0xFFFFu) | (1u << 16), // tensor_dim0[31:16] | tensor_dim1=1
               (nd & 0xFFFFu) << 16,                // tensor_dim1 hi=0 | tile_dim0=nd
               1u,                                  // tile_dim1=1, tile_dim2=0
               nd,                                  // tensor_dim0_stride[31:0]
               0u, 0u };                            // stride hi / dim1_stride
  u32x4 zz = { 0u, 0u, 0u, 0u };
  asm volatile("tensor_load_to_lds %0, %1, %2, %3"
               :: "s"(g0), "s"(g1), "s"(zz), "s"(zz)
               : "memory");
}

// ---------------- weight repack: f32 (out_d x in_d) -> bf16 WMMA-B fragments ----------------
__global__ void pack_b(const float* __restrict__ W, __bf16* __restrict__ P,
                       int out_d, int in_d, int NT, int KT) {
  int idx = blockIdx.x * blockDim.x + threadIdx.x;
  int total = NT * KT * 512;
  if (idx >= total) return;
  int e    = idx & 15;
  int lane = (idx >> 4) & 31;
  int frag = idx >> 9;
  int kt = frag % KT;
  int nt = frag / KT;
  int n = nt * 16 + (lane & 15);
  int k = kt * 32 + ((lane & 16) ? 8 : 0) + ((e < 8) ? e : (8 + e));
  float v = (n < out_d && k < in_d) ? W[(size_t)n * in_d + k] : 0.f;
  P[idx] = (__bf16)v;
}

// ---------------- GRU encoder + projection ----------------
// LDS (block): [0,96K) whh | [96K,120K) wih | [120K,136K) proj | per-wave 12KB h state
#define ENC_OFF_WIH   98304
#define ENC_OFF_PROJ  122880
#define ENC_OFF_WAVE  139264
#define ENC_WAVE_B    12288
#define ENC_SMEM      (ENC_OFF_WAVE + 2 * ENC_WAVE_B)   // 163840 B = 160KB

__global__ __launch_bounds__(64) void enc_kernel(
    const float* __restrict__ cspd,      // (B,T,5,5)
    const __bf16* __restrict__ wihp_g,   // [24][1][512]
    const __bf16* __restrict__ whhp_g,   // [24][4][512]
    const __bf16* __restrict__ projp_g,  // [4][8][512]
    const float* __restrict__ gbias,     // 384
    const float* __restrict__ gbias_n,   // 128
    const float* __restrict__ projb,     // 64
    float* __restrict__ ctxbuf)          // (B,64)
{
  extern __shared__ char smem[];
  const int wave = threadIdx.x >> 5;
  const int lane = threadIdx.x & 31;

  if (wave == 0) {
    tdm_load_1d(whhp_g,  smem,                 24 * 4 * 512 * 2);
    tdm_load_1d(wihp_g,  smem + ENC_OFF_WIH,   24 * 1 * 512 * 2);
    tdm_load_1d(projp_g, smem + ENC_OFF_PROJ,  4  * 8 * 512 * 2);
    __builtin_amdgcn_s_wait_tensorcnt(0);
  }
  const __bf16* whhp  = (const __bf16*)(smem);
  const __bf16* wihp  = (const __bf16*)(smem + ENC_OFF_WIH);
  const __bf16* projp = (const __bf16*)(smem + ENC_OFF_PROJ);
  __bf16* hbf = (__bf16*)(smem + ENC_OFF_WAVE + wave * ENC_WAVE_B);
  float*  hf  = (float*)(smem + ENC_OFF_WAVE + wave * ENC_WAVE_B + 4096);

  const int tile = blockIdx.x * 2 + wave;     // 0..255
  const int b0   = tile * 16;
  const int m    = lane & 15;
  const int ln   = lane & 15;
  const int mb   = (lane >> 4) * 8;
  const int kb   = (lane & 16) ? 8 : 0;

  for (int i = lane; i < 16 * HID; i += 32) { hbf[i] = (__bf16)0.f; hf[i] = 0.f; }
  __syncthreads();   // weights staged + h initialized

  for (int t = 0; t < T_STEPS; t++) {
    const float* cb = cspd + ((size_t)(b0 + m) * T_STEPS + t) * SDIM;
    if (t + 1 < T_STEPS) __builtin_prefetch(cb + SDIM, 0, 0);

    // x A-fragment: symmetrized flattened SPD, K padded 25 -> 32
    v16bf xa;
#pragma unroll
    for (int e = 0; e < 16; e++) {
      int k = kb + ((e < 8) ? e : (16 + (e - 8)));
      float v = 0.f;
      if (k < SDIM) {
        int i5 = k / 5, j5 = k % 5;
        v = 0.5f * (cb[i5 * 5 + j5] + cb[j5 * 5 + i5]);
      }
      xa[e] = (__bf16)v;
    }
    // previous h as 4 A-fragments (captured before overwrite this step)
    v16bf ha[4];
#pragma unroll
    for (int kt = 0; kt < 4; kt++) ha[kt] = load_a_frag(hbf, HID, kt);

    for (int j = 0; j < 8; j++) {           // hidden 16-wide tile
      float bR = gbias[       j * 16 + ln];
      float bZ = gbias[128 + j * 16 + ln];
      float bG = gbias[256 + j * 16 + ln];
      v8f aR, aZ, aG, aH;
#pragma unroll
      for (int r = 0; r < 8; r++) { aR[r] = bR; aZ[r] = bZ; aG[r] = bG; aH[r] = 0.f; }
      aR = wmma_bf16(xa, load_b_frag(wihp, j,      0, 1), aR);
      aZ = wmma_bf16(xa, load_b_frag(wihp, j + 8,  0, 1), aZ);
      aG = wmma_bf16(xa, load_b_frag(wihp, j + 16, 0, 1), aG);
#pragma unroll
      for (int kt = 0; kt < 4; kt++) {
        aR = wmma_bf16(ha[kt], load_b_frag(whhp, j,      kt, 4), aR);
        aZ = wmma_bf16(ha[kt], load_b_frag(whhp, j + 8,  kt, 4), aZ);
        aH = wmma_bf16(ha[kt], load_b_frag(whhp, j + 16, kt, 4), aH);
      }
      float bn = gbias_n[j * 16 + ln];
#pragma unroll
      for (int r = 0; r < 8; r++) {
        int row = mb + r;
        float hold = hf[row * HID + j * 16 + ln];
        float rr = sigmoidf_(aR[r]);
        float zz = sigmoidf_(aZ[r]);
        float gg = tanhf_(aG[r] + rr * (aH[r] + bn));
        float hn = (1.f - zz) * gg + zz * hold;
        hf[row * HID + j * 16 + ln]  = hn;
        hbf[row * HID + j * 16 + ln] = (__bf16)hn;
      }
    }
  }

  // ctx = hT @ proj_w^T + proj_b
#pragma unroll
  for (int nt = 0; nt < 4; nt++) {
    float bv = projb[nt * 16 + ln];
    v8f acc;
#pragma unroll
    for (int r = 0; r < 8; r++) acc[r] = bv;
#pragma unroll
    for (int kt = 0; kt < 8; kt++)
      acc = wmma_bf16(load_a_frag(hbf, HID, kt), load_b_frag(projp, nt, kt, 8), acc);
#pragma unroll
    for (int r = 0; r < 8; r++)
      ctxbuf[(size_t)(b0 + mb + r) * CDIM + nt * 16 + ln] = acc[r];
  }
}

// ---------------- MLP building blocks ----------------
template<int KT, int LD>
__device__ __forceinline__ void mlp8(const __bf16* inbuf, const __bf16* wp,
                                     const float* bias, __bf16* outbuf) {
  const int lane = threadIdx.x & 31;
  const int ln = lane & 15;
  const int mb = (lane >> 4) * 8;
  for (int nt = 0; nt < 8; nt++) {
    float bv = bias[nt * 16 + ln];
    v8f acc;
#pragma unroll
    for (int r = 0; r < 8; r++) acc[r] = bv;
#pragma unroll
    for (int kt = 0; kt < KT; kt++)
      acc = wmma_bf16(load_a_frag(inbuf, LD, kt), load_b_frag(wp, nt, kt, KT), acc);
#pragma unroll
    for (int r = 0; r < 8; r++)
      outbuf[(mb + r) * HID + nt * 16 + ln] = (__bf16)siluf_(acc[r]);
  }
}

template<int KT>
__device__ __forceinline__ void head16(const __bf16* inbuf, const __bf16* wp,
                                       const float* bias, int nbias, float* outv) {
  const int lane = threadIdx.x & 31;
  const int ln = lane & 15;
  const int mb = (lane >> 4) * 8;
  float bv = (ln < nbias) ? bias[ln] : 0.f;
  v8f acc;
#pragma unroll
  for (int r = 0; r < 8; r++) acc[r] = bv;
#pragma unroll
  for (int kt = 0; kt < KT; kt++)
    acc = wmma_bf16(load_a_frag(inbuf, HID, kt), load_b_frag(wp, 0, kt, KT), acc);
#pragma unroll
  for (int r = 0; r < 8; r++) outv[(mb + r) * 16 + ln] = acc[r];
}

// ---------------- 5x5 symmetric eigen machinery (per-lane, registers) ----------------
__device__ __forceinline__ void jacobi5(float A[5][5], float V[5][5]) {
#pragma unroll
  for (int i = 0; i < 5; i++)
#pragma unroll
    for (int j = 0; j < 5; j++) V[i][j] = (i == j) ? 1.f : 0.f;
  for (int sweep = 0; sweep < 6; sweep++) {
#pragma unroll
    for (int p = 0; p < 4; p++) {
#pragma unroll
      for (int q = p + 1; q < 5; q++) {
        float apq = A[p][q];
        float theta = 0.5f * (A[q][q] - A[p][p]) * frcp_(apq + copysignf(1e-30f, apq));
        float tt = copysignf(1.f, theta) * frcp_(fabsf(theta) + fsqrt_(theta * theta + 1.f));
        tt = (fabsf(apq) < 1e-12f) ? 0.f : tt;
        float c = rsqrtf(tt * tt + 1.f);
        float s = tt * c;
#pragma unroll
        for (int k = 0; k < 5; k++) {
          float akp = A[k][p], akq = A[k][q];
          A[k][p] = c * akp - s * akq;
          A[k][q] = s * akp + c * akq;
        }
#pragma unroll
        for (int k = 0; k < 5; k++) {
          float apk = A[p][k], aqk = A[q][k];
          A[p][k] = c * apk - s * aqk;
          A[q][k] = s * apk + c * aqk;
        }
#pragma unroll
        for (int k = 0; k < 5; k++) {
          float vkp = V[k][p], vkq = V[k][q];
          V[k][p] = c * vkp - s * vkq;
          V[k][q] = s * vkp + c * vkq;
        }
      }
    }
  }
}

__device__ __forceinline__ void spd_func(const float S[5][5], float Out[5][5], int isExp) {
  float A[5][5], V[5][5];
#pragma unroll
  for (int i = 0; i < 5; i++)
#pragma unroll
    for (int j = 0; j < 5; j++) A[i][j] = S[i][j];
  jacobi5(A, V);
  float f[5];
#pragma unroll
  for (int i = 0; i < 5; i++) {
    float w = A[i][i];
    f[i] = isExp ? __expf(w) : fsqrt_(fmaxf(w, 1e-8f));
  }
#pragma unroll
  for (int i = 0; i < 5; i++)
#pragma unroll
    for (int j = 0; j < 5; j++) {
      float s = 0.f;
#pragma unroll
      for (int k = 0; k < 5; k++) s += V[i][k] * f[k] * V[j][k];
      Out[i][j] = s;
    }
}

__device__ __forceinline__ void mm5(const float A[5][5], const float B[5][5], float C[5][5]) {
#pragma unroll
  for (int i = 0; i < 5; i++)
#pragma unroll
    for (int j = 0; j < 5; j++) {
      float s = 0.f;
#pragma unroll
      for (int k = 0; k < 5; k++) s += A[i][k] * B[k][j];
      C[i][j] = s;
    }
}

// ---------------- SDE integrator ----------------
// LDS (block): staged MLP weights (152KB) then per-wave working set (19200B x 2)
#define SD_OFF_DR0   0        // 24576
#define SD_OFF_DR1   24576    // 32768
#define SD_OFF_DR2   57344    // 32768
#define SD_OFF_DR3   90112    // 4096
#define SD_OFF_DI0   94208    // 24576
#define SD_OFF_DI1   118784   // 32768
#define SD_OFF_DI2   151552   // 4096
#define SD_OFF_WAVE  155648
// per-wave working-set offsets (bytes, relative to wave base)
#define SW_INP   0      // 16x96  bf16 = 3072
#define SW_ACTA  3072   // 16x128 bf16 = 4096
#define SW_ACTB  7168   // 16x128 bf16 = 4096
#define SW_SIG   11264  // 16x25  f32  = 1600
#define SW_CTX   12864  // 16x64  f32  = 4096
#define SW_DV    16960  // 16x16  f32  = 1024
#define SW_SV    17984  // 16x16  f32  = 1024
#define SW_BYTES 19200
#define SDE_SMEM (SD_OFF_WAVE + 2 * SW_BYTES)   // 194048 B

__global__ __launch_bounds__(64) void sde_kernel(
    const float* __restrict__ cspd,
    const float* __restrict__ dWin,
    const float* __restrict__ ctxbuf,
    const __bf16* __restrict__ dr0g, const float* __restrict__ db0,
    const __bf16* __restrict__ dr1g, const float* __restrict__ db1,
    const __bf16* __restrict__ dr2g, const float* __restrict__ db2,
    const __bf16* __restrict__ dr3g, const float* __restrict__ db3,
    const __bf16* __restrict__ di0g, const float* __restrict__ eb0,
    const __bf16* __restrict__ di1g, const float* __restrict__ eb1,
    const __bf16* __restrict__ di2g, const float* __restrict__ eb2,
    float* __restrict__ outp)
{
  extern __shared__ char smem[];
  const int wave = threadIdx.x >> 5;
  const int lane = threadIdx.x & 31;

  if (wave == 0) {
    tdm_load_1d(dr0g, smem + SD_OFF_DR0, 24576);
    tdm_load_1d(dr1g, smem + SD_OFF_DR1, 32768);
    tdm_load_1d(dr2g, smem + SD_OFF_DR2, 32768);
    tdm_load_1d(dr3g, smem + SD_OFF_DR3, 4096);
    tdm_load_1d(di0g, smem + SD_OFF_DI0, 24576);
    tdm_load_1d(di1g, smem + SD_OFF_DI1, 32768);
    tdm_load_1d(di2g, smem + SD_OFF_DI2, 4096);
    __builtin_amdgcn_s_wait_tensorcnt(0);
  }
  const __bf16* dr0p = (const __bf16*)(smem + SD_OFF_DR0);
  const __bf16* dr1p = (const __bf16*)(smem + SD_OFF_DR1);
  const __bf16* dr2p = (const __bf16*)(smem + SD_OFF_DR2);
  const __bf16* dr3p = (const __bf16*)(smem + SD_OFF_DR3);
  const __bf16* di0p = (const __bf16*)(smem + SD_OFF_DI0);
  const __bf16* di1p = (const __bf16*)(smem + SD_OFF_DI1);
  const __bf16* di2p = (const __bf16*)(smem + SD_OFF_DI2);

  char* wbase = smem + SD_OFF_WAVE + wave * SW_BYTES;
  __bf16* inp  = (__bf16*)(wbase + SW_INP);
  __bf16* actA = (__bf16*)(wbase + SW_ACTA);
  __bf16* actB = (__bf16*)(wbase + SW_ACTB);
  float*  sig  = (float*)(wbase + SW_SIG);
  float*  ctxf = (float*)(wbase + SW_CTX);
  float*  dv   = (float*)(wbase + SW_DV);
  float*  sv   = (float*)(wbase + SW_SV);
  const int tile = blockIdx.x * 2 + wave;   // 0..255
  const int b0 = tile * 16;
  const int m  = lane & 15;

  for (int i = lane; i < 16 * CDIM; i += 32)
    ctxf[i] = ctxbuf[(size_t)(b0 + i / CDIM) * CDIM + (i % CDIM)];
  for (int i = lane; i < 16 * SDIM; i += 32) {
    int r = i / SDIM, k = i % SDIM, ii = k / 5, jj = k % 5;
    const float* cs = cspd + ((size_t)(b0 + r) * T_STEPS + (T_STEPS - 1)) * SDIM;
    sig[i] = 0.5f * (cs[ii * 5 + jj] + cs[jj * 5 + ii]);
  }
  // inp = [state | ctx | pad], ctx part constant over steps
  for (int i = lane; i < 16 * 96; i += 32) {
    int r = i / 96, c = i % 96;
    float v = (c >= SDIM && c < SDIM + CDIM) ? ctxf[r * CDIM + (c - SDIM)] : 0.f;
    inp[i] = (__bf16)v;
  }
  __syncthreads();   // weights staged (wave 0) before anyone reads them

  for (int step = 0; step < NSTEPS; step++) {
    for (int i = lane; i < 16 * SDIM; i += 32)
      inp[(i / SDIM) * 96 + (i % SDIM)] = (__bf16)sig[i];

    // drift MLP (4 linears, silu on hidden)
    mlp8<3, 96>(inp,  dr0p, db0, actA);
    mlp8<4, HID>(actA, dr1p, db1, actB);
    mlp8<4, HID>(actB, dr2p, db2, actA);
    head16<4>(actA, dr3p, db3, DDIM, dv);
    // diffusion MLP (3 linears)
    mlp8<3, 96>(inp,  di0p, eb0, actB);
    mlp8<4, HID>(actB, di1p, eb1, actA);
    head16<4>(actA, di2p, eb2, DDIM, sv);

    // geometric exp-map update: lanes 0..15 each own one batch row
    if (lane < 16) {
      const float* dwp = dWin + ((size_t)(b0 + m) * NSTEPS + step) * DDIM;
      float co[DDIM];
#pragma unroll
      for (int k = 0; k < DDIM; k++)
        co[k] = dv[m * 16 + k] * DT_C + softplusf_(sv[m * 16 + k]) * dwp[k];
      float a[5][5];
#pragma unroll
      for (int i = 0; i < 5; i++) a[i][i] = co[i];
      {
        int p = 5;
#pragma unroll
        for (int i = 0; i < 5; i++)
#pragma unroll
          for (int j = i + 1; j < 5; j++) {
            float v = co[p] * 0.70710678118654752f;
            a[i][j] = v; a[j][i] = v; p++;
          }
      }
      float Sg[5][5];
#pragma unroll
      for (int i = 0; i < 5; i++)
#pragma unroll
        for (int j = 0; j < 5; j++) Sg[i][j] = sig[m * SDIM + i * 5 + j];
      float L[5][5], E[5][5], Tm[5][5], Sn[5][5];
      spd_func(Sg, L, 0);   // symmetric sqrt
      spd_func(a,  E, 1);   // symmetric expm
      mm5(L, E, Tm);
      mm5(Tm, L, Sn);
#pragma unroll
      for (int i = 0; i < 5; i++)
#pragma unroll
        for (int j = 0; j < 5; j++)
          sig[m * SDIM + i * 5 + j] = 0.5f * (Sn[i][j] + Sn[j][i]);
    }
  }

  if (lane < 16) {
#pragma unroll
    for (int k = 0; k < SDIM; k++)
      outp[(size_t)(b0 + m) * SDIM + k] = sig[m * SDIM + k];
  }
}

// ---------------- launch ----------------
extern "C" void kernel_launch(void* const* d_in, const int* in_sizes, int n_in,
                              void* d_out, int out_size, void* d_ws, size_t ws_size,
                              hipStream_t stream) {
  (void)in_sizes; (void)n_in; (void)out_size; (void)ws_size;
  const float* cspd   = (const float*)d_in[0];
  const float* dWin   = (const float*)d_in[1];
  const float* wih    = (const float*)d_in[2];
  const float* whh    = (const float*)d_in[3];
  const float* gbias  = (const float*)d_in[4];
  const float* gbiasn = (const float*)d_in[5];
  const float* projw  = (const float*)d_in[6];
  const float* projb  = (const float*)d_in[7];
  const float* dw0 = (const float*)d_in[8];  const float* db0 = (const float*)d_in[9];
  const float* dw1 = (const float*)d_in[10]; const float* db1 = (const float*)d_in[11];
  const float* dw2 = (const float*)d_in[12]; const float* db2 = (const float*)d_in[13];
  const float* dw3 = (const float*)d_in[14]; const float* db3 = (const float*)d_in[15];
  const float* ew0 = (const float*)d_in[16]; const float* eb0 = (const float*)d_in[17];
  const float* ew1 = (const float*)d_in[18]; const float* eb1 = (const float*)d_in[19];
  const float* ew2 = (const float*)d_in[20]; const float* eb2 = (const float*)d_in[21];

  size_t off = 0;
  auto carve = [&](size_t bytes) -> char* {
    char* p = (char*)d_ws + off;
    off += (bytes + 255) & ~(size_t)255;
    return p;
  };
  float*  ctxbuf = (float*)carve((size_t)B_TOTAL * CDIM * sizeof(float));
  __bf16* wihp  = (__bf16*)carve((size_t)24 * 1 * 512 * 2);
  __bf16* whhp  = (__bf16*)carve((size_t)24 * 4 * 512 * 2);
  __bf16* projp = (__bf16*)carve((size_t)4  * 8 * 512 * 2);
  __bf16* dr0p  = (__bf16*)carve((size_t)8  * 3 * 512 * 2);
  __bf16* dr1p  = (__bf16*)carve((size_t)8  * 4 * 512 * 2);
  __bf16* dr2p  = (__bf16*)carve((size_t)8  * 4 * 512 * 2);
  __bf16* dr3p  = (__bf16*)carve((size_t)1  * 4 * 512 * 2);
  __bf16* di0p  = (__bf16*)carve((size_t)8  * 3 * 512 * 2);
  __bf16* di1p  = (__bf16*)carve((size_t)8  * 4 * 512 * 2);
  __bf16* di2p  = (__bf16*)carve((size_t)1  * 4 * 512 * 2);

  auto pack = [&](const float* W, __bf16* P, int od, int id, int NT, int KT) {
    int total = NT * KT * 512;
    pack_b<<<(total + 255) / 256, 256, 0, stream>>>(W, P, od, id, NT, KT);
  };
  pack(wih,   wihp,  384, 25,  24, 1);
  pack(whh,   whhp,  384, 128, 24, 4);
  pack(projw, projp, 64,  128, 4,  8);
  pack(dw0,   dr0p,  128, 89,  8,  3);
  pack(dw1,   dr1p,  128, 128, 8,  4);
  pack(dw2,   dr2p,  128, 128, 8,  4);
  pack(dw3,   dr3p,  15,  128, 1,  4);
  pack(ew0,   di0p,  128, 89,  8,  3);
  pack(ew1,   di1p,  128, 128, 8,  4);
  pack(ew2,   di2p,  15,  128, 1,  4);

  // allow >64KB dynamic LDS (320KB per WGP on CDNA5); host-side, capture-safe
  (void)hipFuncSetAttribute((const void*)enc_kernel,
                            hipFuncAttributeMaxDynamicSharedMemorySize, ENC_SMEM);
  (void)hipFuncSetAttribute((const void*)sde_kernel,
                            hipFuncAttributeMaxDynamicSharedMemorySize, SDE_SMEM);

  // 256 batch tiles, 2 waves/block -> 128 blocks each
  enc_kernel<<<128, 64, ENC_SMEM, stream>>>(cspd, wihp, whhp, projp,
                                            gbias, gbiasn, projb, ctxbuf);
  sde_kernel<<<128, 64, SDE_SMEM, stream>>>(cspd, dWin, ctxbuf,
                                            dr0p, db0, dr1p, db1, dr2p, db2, dr3p, db3,
                                            di0p, eb0, di1p, eb1, di2p, eb2,
                                            (float*)d_out);
}
